// SimpleMOE_37503654429096
// MI455X (gfx1250) — compile-verified, matrix operand order checked by reading, not run on hardware
//
#include <hip/hip_runtime.h>
#include <hip/hip_bf16.h>

// Problem constants (match reference)
#define BB 4096
#define DD 2048
#define HH 8192
#define EE 8
#define KK 2

typedef __attribute__((ext_vector_type(16))) __bf16 v16bf;
typedef __attribute__((ext_vector_type(8)))  float  v8f;

union Frag16 { v16bf v; unsigned long long u[4]; };
union Pack4  { unsigned long long u; __bf16 h[4]; };
union Pack2  { unsigned int u; __bf16 h[2]; };

// ---------------- workspace layout (bytes) ----------------
#define OFF_COUNTS 0                              // E ints
#define OFF_RP     1024                           // B*E floats   (131072 B)
#define OFF_TOK    (OFF_RP + BB*EE*4)             // E*B ints     (131072 B)
#define OFF_SLOT   (OFF_TOK + EE*BB*4)            // E*B ints     (131072 B)
#define OFF_CW     (OFF_SLOT + EE*BB*4)           // B*K floats   (32768 B)
#define OFF_H      (OFF_CW + BB*KK*4)             // B*K*H bf16   (134217728 B)
#define OFF_Y      (OFF_H + (size_t)BB*KK*HH*2)   // B*K*D f32    (67108864 B)

// ---------------- init ----------------
__global__ void moe_init(int* __restrict__ counts) {
    if (threadIdx.x < EE) counts[threadIdx.x] = 0;
}

// ---------------- gating: logits, top-2, router softmax, dispatch ----------------
__global__ __launch_bounds__(256)
void moe_gate(const float* __restrict__ x, const float* __restrict__ gw,
              const float* __restrict__ gb, int* __restrict__ counts,
              float* __restrict__ rp, int* __restrict__ tok_list,
              int* __restrict__ slot_list, float* __restrict__ cw_slot)
{
    const int wave  = threadIdx.x >> 5;
    const int lane  = threadIdx.x & 31;
    const int token = blockIdx.x * 8 + wave;

    float acc[EE];
#pragma unroll
    for (int e = 0; e < EE; ++e) acc[e] = 0.f;

    const float* xr = x + (size_t)token * DD;
    for (int d = lane; d < DD; d += 32) {
        const float xv = xr[d];
        const float* g = gw + (size_t)d * EE;
#pragma unroll
        for (int e = 0; e < EE; ++e) acc[e] += xv * g[e];
    }
#pragma unroll
    for (int e = 0; e < EE; ++e) {
#pragma unroll
        for (int off = 16; off > 0; off >>= 1)
            acc[e] += __shfl_xor(acc[e], off);
    }

    if (lane == 0) {
        float lg[EE];
#pragma unroll
        for (int e = 0; e < EE; ++e) lg[e] = acc[e] + gb[e];

        int i0 = 0;
#pragma unroll
        for (int e = 1; e < EE; ++e) if (lg[e] > lg[i0]) i0 = e;
        int i1 = (i0 == 0) ? 1 : 0;
#pragma unroll
        for (int e = 0; e < EE; ++e) if (e != i0 && lg[e] > lg[i1]) i1 = e;

        const float m  = lg[i0];
        const float x1 = __expf(lg[i1] - m);
        const float iv = 1.f / (1.f + x1);
        const float cw0 = iv, cw1 = x1 * iv;

        float s = 0.f, r[EE];
#pragma unroll
        for (int e = 0; e < EE; ++e) { r[e] = __expf(lg[e] - m); s += r[e]; }
        const float is = 1.f / s;
#pragma unroll
        for (int e = 0; e < EE; ++e) rp[token * EE + e] = r[e] * is;

        int p0 = atomicAdd(&counts[i0], 1);
        tok_list[i0 * BB + p0]  = token;
        slot_list[i0 * BB + p0] = token * KK + 0;
        cw_slot[token * KK + 0] = cw0;

        int p1 = atomicAdd(&counts[i1], 1);
        tok_list[i1 * BB + p1]  = token;
        slot_list[i1 * BB + p1] = token * KK + 1;
        cw_slot[token * KK + 1] = cw1;
    }
}

// ---------------- fragment load helpers (ISA 16-bit A/B VGPR layout) ----------------
// lane (l): m/n = l&15, K offsets {0,4,16,20} + 8*(l>>4), 4 contiguous bf16 each.
__device__ __forceinline__ v16bf load_frag(const __bf16* p) {
    Frag16 f;
    f.u[0] = *(const unsigned long long*)(p + 0);
    f.u[1] = *(const unsigned long long*)(p + 4);
    f.u[2] = *(const unsigned long long*)(p + 16);
    f.u[3] = *(const unsigned long long*)(p + 20);
    return f.v;
}

// ---------------- fc1: h = relu(x @ w1[e] + b1[e]), bf16 WMMA, BM=32 ----------------
__global__ __launch_bounds__(256)
void moe_fc1(const float* __restrict__ x, const float* __restrict__ w1,
             const float* __restrict__ b1, const int* __restrict__ counts,
             const int* __restrict__ tok_list, const int* __restrict__ slot_list,
             __bf16* __restrict__ h)
{
    const int e   = blockIdx.z;
    const int cnt = counts[e];
    const int m0  = blockIdx.x * 32;
    if (m0 >= cnt) return;
    const int rows = min(32, cnt - m0);
    const int n0   = blockIdx.y * 128;
    const int tid  = threadIdx.x;

    __shared__ int s_tok[32], s_slot[32];
    __shared__ __align__(16) __bf16 sA[32 * 64];     // [m][k]  4 KB
    __shared__ __align__(16) __bf16 sBT[128 * 64];   // [n][k] 16 KB
    __shared__ __align__(16) float  sC[32 * 128];    //        16 KB

    if (tid < 32) {
        int idx     = min(m0 + tid, cnt - 1);
        s_tok[tid]  = tok_list[e * BB + idx];
        s_slot[tid] = slot_list[e * BB + idx];
    }
    __syncthreads();

    const int wv = tid >> 5, lane = tid & 31;
    const int lm = lane & 15, lhi = lane >> 4;
    const float* w1e = w1 + (size_t)e * DD * HH;

    const int ar  = tid >> 3, akq = (tid & 7) * 8;      // A: row 0..31, 8 k each
    const int bkp = (tid >> 3) * 2, bnq = (tid & 7) * 16; // B: k-pair, 16 n each
    const size_t arow = (size_t)s_tok[ar] * DD;

    v8f c0 = {}, c1 = {};
    for (int kb = 0; kb < DD; kb += 64) {
        { // stage A (32 x 64), fp32 -> bf16
            const float4 f0 = *(const float4*)(x + arow + kb + akq);
            const float4 f1 = *(const float4*)(x + arow + kb + akq + 4);
            Pack4 p0, p1;
            p0.h[0] = (__bf16)f0.x; p0.h[1] = (__bf16)f0.y;
            p0.h[2] = (__bf16)f0.z; p0.h[3] = (__bf16)f0.w;
            p1.h[0] = (__bf16)f1.x; p1.h[1] = (__bf16)f1.y;
            p1.h[2] = (__bf16)f1.z; p1.h[3] = (__bf16)f1.w;
            *(unsigned long long*)&sA[ar * 64 + akq]     = p0.u;
            *(unsigned long long*)&sA[ar * 64 + akq + 4] = p1.u;
        }
        { // stage B transposed (64k x 128n -> [n][k]), paired-k b32 stores
            const float* s0 = w1e + (size_t)(kb + bkp) * HH + n0 + bnq;
            const float* s1 = s0 + HH;
#pragma unroll
            for (int q = 0; q < 4; ++q) {
                const float4 f0 = *(const float4*)(s0 + q * 4);
                const float4 f1 = *(const float4*)(s1 + q * 4);
                const int nb = bnq + q * 4;
                Pack2 p;
                p.h[0]=(__bf16)f0.x; p.h[1]=(__bf16)f1.x; *(unsigned*)&sBT[(nb+0)*64 + bkp] = p.u;
                p.h[0]=(__bf16)f0.y; p.h[1]=(__bf16)f1.y; *(unsigned*)&sBT[(nb+1)*64 + bkp] = p.u;
                p.h[0]=(__bf16)f0.z; p.h[1]=(__bf16)f1.z; *(unsigned*)&sBT[(nb+2)*64 + bkp] = p.u;
                p.h[0]=(__bf16)f0.w; p.h[1]=(__bf16)f1.w; *(unsigned*)&sBT[(nb+3)*64 + bkp] = p.u;
            }
        }
        __syncthreads();
#pragma unroll
        for (int ks = 0; ks < 64; ks += 32) {
            const int kbase = ks + lhi * 8;
            const v16bf a0 = load_frag(&sA[lm * 64 + kbase]);
            const v16bf a1 = load_frag(&sA[(16 + lm) * 64 + kbase]);
            const v16bf b  = load_frag(&sBT[(wv * 16 + lm) * 64 + kbase]);
            c0 = __builtin_amdgcn_wmma_f32_16x16x32_bf16(false, a0, false, b,
                                                         (short)0, c0, false, false);
            c1 = __builtin_amdgcn_wmma_f32_16x16x32_bf16(false, a1, false, b,
                                                         (short)0, c1, false, false);
        }
        __syncthreads();
    }

    { // bias + relu -> sC (transpose through LDS)
        const float bias = b1[e * HH + n0 + wv * 16 + lm];
#pragma unroll
        for (int r = 0; r < 8; ++r) {
            float v0 = c0[r] + bias; v0 = v0 > 0.f ? v0 : 0.f;
            float v1 = c1[r] + bias; v1 = v1 > 0.f ? v1 : 0.f;
            sC[(r + lhi * 8) * 128 + wv * 16 + lm]        = v0;
            sC[(16 + r + lhi * 8) * 128 + wv * 16 + lm]   = v1;
        }
    }
    __syncthreads();
    { // coalesced bf16 store of h rows by slot (16 elems / thread)
        const int r = tid >> 3, j0 = (tid & 7) * 16;
        if (r < rows) {
            __bf16* dst = h + (size_t)s_slot[r] * HH + n0 + j0;
#pragma unroll
            for (int q = 0; q < 4; ++q) {
                Pack4 p;
#pragma unroll
                for (int i = 0; i < 4; ++i)
                    p.h[i] = (__bf16)sC[r * 128 + j0 + q * 4 + i];
                *(unsigned long long*)(dst + q * 4) = p.u;
            }
        }
    }
}

// ---------------- fc2: y = h @ w2[e] + b2[e], bf16 WMMA, BM=32 ----------------
__global__ __launch_bounds__(256)
void moe_fc2(const __bf16* __restrict__ h, const float* __restrict__ w2,
             const float* __restrict__ b2, const int* __restrict__ counts,
             const int* __restrict__ slot_list, float* __restrict__ y)
{
    const int e   = blockIdx.z;
    const int cnt = counts[e];
    const int m0  = blockIdx.x * 32;
    if (m0 >= cnt) return;
    const int rows = min(32, cnt - m0);
    const int n0   = blockIdx.y * 128;
    const int tid  = threadIdx.x;

    __shared__ int s_slot[32];
    __shared__ __align__(16) __bf16 sA[32 * 64];
    __shared__ __align__(16) __bf16 sBT[128 * 64];
    __shared__ __align__(16) float  sC[32 * 128];

    if (tid < 32) {
        int idx     = min(m0 + tid, cnt - 1);
        s_slot[tid] = slot_list[e * BB + idx];
    }
    __syncthreads();

    const int wv = tid >> 5, lane = tid & 31;
    const int lm = lane & 15, lhi = lane >> 4;
    const float* w2e = w2 + (size_t)e * HH * DD;

    const int ar  = tid >> 3, akq = (tid & 7) * 8;
    const int bkp = (tid >> 3) * 2, bnq = (tid & 7) * 16;
    const size_t arow = (size_t)s_slot[ar] * HH;

    v8f c0 = {}, c1 = {};
    for (int kb = 0; kb < HH; kb += 64) {
        { // stage A (bf16 already): straight 16B copy
            *(uint4*)&sA[ar * 64 + akq] = *(const uint4*)(h + arow + kb + akq);
        }
        { // stage B transposed, paired-k b32 stores
            const float* s0 = w2e + (size_t)(kb + bkp) * DD + n0 + bnq;
            const float* s1 = s0 + DD;
#pragma unroll
            for (int q = 0; q < 4; ++q) {
                const float4 f0 = *(const float4*)(s0 + q * 4);
                const float4 f1 = *(const float4*)(s1 + q * 4);
                const int nb = bnq + q * 4;
                Pack2 p;
                p.h[0]=(__bf16)f0.x; p.h[1]=(__bf16)f1.x; *(unsigned*)&sBT[(nb+0)*64 + bkp] = p.u;
                p.h[0]=(__bf16)f0.y; p.h[1]=(__bf16)f1.y; *(unsigned*)&sBT[(nb+1)*64 + bkp] = p.u;
                p.h[0]=(__bf16)f0.z; p.h[1]=(__bf16)f1.z; *(unsigned*)&sBT[(nb+2)*64 + bkp] = p.u;
                p.h[0]=(__bf16)f0.w; p.h[1]=(__bf16)f1.w; *(unsigned*)&sBT[(nb+3)*64 + bkp] = p.u;
            }
        }
        __syncthreads();
#pragma unroll
        for (int ks = 0; ks < 64; ks += 32) {
            const int kbase = ks + lhi * 8;
            const v16bf a0 = load_frag(&sA[lm * 64 + kbase]);
            const v16bf a1 = load_frag(&sA[(16 + lm) * 64 + kbase]);
            const v16bf b  = load_frag(&sBT[(wv * 16 + lm) * 64 + kbase]);
            c0 = __builtin_amdgcn_wmma_f32_16x16x32_bf16(false, a0, false, b,
                                                         (short)0, c0, false, false);
            c1 = __builtin_amdgcn_wmma_f32_16x16x32_bf16(false, a1, false, b,
                                                         (short)0, c1, false, false);
        }
        __syncthreads();
    }

    { // + bias -> sC
        const float bias = b2[e * DD + n0 + wv * 16 + lm];
#pragma unroll
        for (int r = 0; r < 8; ++r) {
            sC[(r + lhi * 8) * 128 + wv * 16 + lm]      = c0[r] + bias;
            sC[(16 + r + lhi * 8) * 128 + wv * 16 + lm] = c1[r] + bias;
        }
    }
    __syncthreads();
    { // coalesced fp32 store of y rows by slot (16 elems / thread)
        const int r = tid >> 3, j0 = (tid & 7) * 16;
        if (r < rows) {
            float* dst = y + (size_t)s_slot[r] * DD + n0 + j0;
#pragma unroll
            for (int q = 0; q < 4; ++q)
                *(float4*)(dst + q * 4) = *(const float4*)&sC[r * 128 + j0 + q * 4];
        }
    }
}

// ---------------- combine: out[b] = cw0*y[2b] + cw1*y[2b+1] ----------------
__global__ __launch_bounds__(256)
void moe_combine(const float* __restrict__ y, const float* __restrict__ cw,
                 float* __restrict__ out)
{
    const int b = blockIdx.y;
    const int d = blockIdx.x * 1024 + threadIdx.x * 4;
    const float c0 = cw[b * 2 + 0];
    const float c1 = cw[b * 2 + 1];
    const float4 y0 = *(const float4*)(y + (size_t)(2 * b + 0) * DD + d);
    const float4 y1 = *(const float4*)(y + (size_t)(2 * b + 1) * DD + d);
    float4 o;
    o.x = c0 * y0.x + c1 * y1.x;
    o.y = c0 * y0.y + c1 * y1.y;
    o.z = c0 * y0.z + c1 * y1.z;
    o.w = c0 * y0.w + c1 * y1.w;
    *(float4*)(out + (size_t)b * DD + d) = o;
}

// ---------------- aux loss (deterministic reduction) ----------------
__global__ __launch_bounds__(256)
void moe_aux(const int* __restrict__ counts, const float* __restrict__ rp,
             float* __restrict__ out_aux)
{
    __shared__ float red[256];
    __shared__ float sc[EE];
    if (threadIdx.x < EE) sc[threadIdx.x] = (float)counts[threadIdx.x];
    __syncthreads();
    float s = 0.f;
    for (int b = threadIdx.x; b < BB; b += 256) {
        const float* r = rp + b * EE;
#pragma unroll
        for (int e = 0; e < EE; ++e) s += sc[e] * r[e];
    }
    red[threadIdx.x] = s;
    __syncthreads();
    for (int off = 128; off > 0; off >>= 1) {
        if (threadIdx.x < off) red[threadIdx.x] += red[threadIdx.x + off];
        __syncthreads();
    }
    if (threadIdx.x == 0)
        out_aux[0] = (float)EE * red[0] / ((float)BB * (float)BB);
}

// ---------------- launch ----------------
extern "C" void kernel_launch(void* const* d_in, const int* in_sizes, int n_in,
                              void* d_out, int out_size, void* d_ws, size_t ws_size,
                              hipStream_t stream)
{
    (void)in_sizes; (void)n_in; (void)out_size; (void)ws_size;
    const float* x      = (const float*)d_in[0];
    const float* gate_w = (const float*)d_in[1];
    const float* gate_b = (const float*)d_in[2];
    const float* w1     = (const float*)d_in[3];
    const float* b1     = (const float*)d_in[4];
    const float* w2     = (const float*)d_in[5];
    const float* b2     = (const float*)d_in[6];
    float* out = (float*)d_out;

    char* ws = (char*)d_ws;
    int*    counts    = (int*)   (ws + OFF_COUNTS);
    float*  rp        = (float*) (ws + OFF_RP);
    int*    tok_list  = (int*)   (ws + OFF_TOK);
    int*    slot_list = (int*)   (ws + OFF_SLOT);
    float*  cw_slot   = (float*) (ws + OFF_CW);
    __bf16* hbuf      = (__bf16*)(ws + OFF_H);
    float*  ybuf      = (float*) (ws + OFF_Y);

    moe_init<<<1, 64, 0, stream>>>(counts);
    moe_gate<<<BB / 8, 256, 0, stream>>>(x, gate_w, gate_b, counts, rp,
                                         tok_list, slot_list, cw_slot);
    moe_fc1<<<dim3(BB / 32, HH / 128, EE), 256, 0, stream>>>(
        x, w1, b1, counts, tok_list, slot_list, hbuf);
    moe_fc2<<<dim3(BB / 32, DD / 128, EE), 256, 0, stream>>>(
        hbuf, w2, b2, counts, slot_list, ybuf);
    moe_combine<<<dim3(DD / 1024, BB), 256, 0, stream>>>(ybuf, cw_slot, out);
    moe_aux<<<1, 256, 0, stream>>>(counts, rp, out + (size_t)BB * DD);
}
